// BiLinearAttention_30897994727724
// MI455X (gfx1250) — matrix operand order repaired and checked by reading, not compiled, not run on hardware
//
#include <hip/hip_runtime.h>
#include <stdint.h>

#define BB    16
#define SCTX  2048
#define SQ    2048
#define DCTX  1024
#define DQ    1024

typedef __attribute__((ext_vector_type(16))) __bf16       v16bf;
typedef __attribute__((ext_vector_type(8)))  __bf16       v8bf;
typedef __attribute__((ext_vector_type(4)))  __bf16       v4bf;
typedef __attribute__((ext_vector_type(8)))  float        v8f;
typedef __attribute__((ext_vector_type(8)))  unsigned int v8ui;

__device__ __forceinline__ __bf16 tobf(float x) { return (__bf16)x; }

// ---- LDS fragment loaders (WMMA bf16 16x16x32 operands, contiguous k order) -------------
// A (16x32, MxK): lane<16 -> K 0..15, lane>=16 -> K 16..31
__device__ __forceinline__ v16bf load_a_frag(const __bf16* At, int mbase, int lane) {
    int m  = mbase + (lane & 15);
    int kb = (lane >> 4) * 16;
    v8bf lo = *(const v8bf*)(At + m * 32 + kb);
    v8bf hi = *(const v8bf*)(At + m * 32 + kb + 8);
    return __builtin_shufflevector(lo, hi, 0,1,2,3,4,5,6,7,8,9,10,11,12,13,14,15);
}
// B (32x16, KxN), LDS holds Bt[n][k] (N-major): lane<16 -> K 0..15, lane>=16 -> K 16..31
__device__ __forceinline__ v16bf load_b_frag(const __bf16* Bt, int nbase, int lane) {
    int n  = nbase + (lane & 15);
    int kb = (lane >> 4) * 16;
    return *(const v16bf*)(Bt + n * 32 + kb);   // 32B contiguous
}

// ---- global->reg->LDS tile movers ----------------------------------------
// A tile: 128 rows x 32 K (f32 source), 256 threads -> 4 float4 each
__device__ __forceinline__ void gloadA(const float* A, int ldA, int k0, int tid, float4 a[4]) {
#pragma unroll
    for (int r = 0; r < 4; ++r) {
        int idx = tid + 256 * r, row = idx >> 3, c4 = idx & 7;
        a[r] = *(const float4*)(A + (size_t)row * ldA + k0 + c4 * 4);
    }
}
__device__ __forceinline__ void storeA(__bf16* At, int tid, const float4 a[4]) {
#pragma unroll
    for (int r = 0; r < 4; ++r) {
        int idx = tid + 256 * r, row = idx >> 3, c4 = idx & 7;
        v4bf t = { tobf(a[r].x), tobf(a[r].y), tobf(a[r].z), tobf(a[r].w) };
        *(v4bf*)(At + row * 32 + c4 * 4) = t;
    }
}
// B tile: 32 K x 256 N (f32 [k][n] source) -> transpose into Bt[n][k]; 8 float4/thread
__device__ __forceinline__ void gloadB32(const float* Bs, int ldB, int k0, int n0, int tid, float4 b[8]) {
#pragma unroll
    for (int r = 0; r < 8; ++r) {
        int idx = tid + 256 * r, k = idx >> 6, c4 = idx & 63;
        b[r] = *(const float4*)(Bs + (size_t)(k0 + k) * ldB + n0 + c4 * 4);
    }
}
__device__ __forceinline__ void storeB32t(__bf16* Bt, int tid, const float4 b[8]) {
#pragma unroll
    for (int r = 0; r < 8; ++r) {
        int idx = tid + 256 * r, k = idx >> 6, c4 = idx & 63;
        Bt[(c4 * 4 + 0) * 32 + k] = tobf(b[r].x);
        Bt[(c4 * 4 + 1) * 32 + k] = tobf(b[r].y);
        Bt[(c4 * 4 + 2) * 32 + k] = tobf(b[r].z);
        Bt[(c4 * 4 + 3) * 32 + k] = tobf(b[r].w);
    }
}
// B tile already bf16 in Bt[n][k] row-major (CW rows): 256 n x 32 k = 16KB, 2x32B/thread
__device__ __forceinline__ void gloadBbf(const __bf16* Bs, int ldB, int k0, int tid, v8ui b[2]) {
#pragma unroll
    for (int r = 0; r < 2; ++r) {
        int idx = tid + 256 * r, n = idx >> 1, half = idx & 1;
        b[r] = *(const v8ui*)(Bs + (size_t)n * ldB + k0 + half * 16);
    }
}
__device__ __forceinline__ void storeBbf(__bf16* Bt, int tid, const v8ui b[2]) {
#pragma unroll
    for (int r = 0; r < 2; ++r) {
        int idx = tid + 256 * r, n = idx >> 1, half = idx & 1;
        *(v8ui*)(Bt + n * 32 + half * 16) = b[r];
    }
}

// ---- per-wave 64x64 WMMA step (16 wmma) ----------------------------------
__device__ __forceinline__ void mma_step(const __bf16* At, const __bf16* Bt,
                                         int wm, int wn, int lane, v8f acc[4][4]) {
    v16bf bf[4];
#pragma unroll
    for (int j = 0; j < 4; ++j) bf[j] = load_b_frag(Bt, wn * 64 + j * 16, lane);
#pragma unroll
    for (int i = 0; i < 4; ++i) {
        v16bf a = load_a_frag(At, wm * 64 + i * 16, lane);
#pragma unroll
        for (int j = 0; j < 4; ++j)
            acc[i][j] = __builtin_amdgcn_wmma_f32_16x16x32_bf16(false, a, false, bf[j],
                                                                (short)0, acc[i][j], false, false);
    }
}

// ============ K1: CW[b,s,e] = context[b,s,:] @ W  (bf16 out) ==============
__global__ __launch_bounds__(256) void k1_cw(const float* __restrict__ ctx,
                                             const float* __restrict__ W,
                                             __bf16* __restrict__ CW) {
    __shared__ __attribute__((aligned(32))) __bf16 At[2][128 * 32];
    __shared__ __attribute__((aligned(32))) __bf16 Bt[2][256 * 32];
    const int tid = threadIdx.x, lane = tid & 31, wave = tid >> 5;
    const int wm = wave & 1, wn = wave >> 1;
    const int m0 = blockIdx.x * 128, n0 = blockIdx.y * 256, b = blockIdx.z;
    const float* A = ctx + (size_t)b * SCTX * DCTX + (size_t)m0 * DCTX;

    v8f acc[4][4] = {};
    float4 ar[4], br[8];
    const int NK = DCTX / 32;
    gloadA(A, DCTX, 0, tid, ar);  gloadB32(W, DQ, 0, n0, tid, br);
    storeA(At[0], tid, ar);       storeB32t(Bt[0], tid, br);
    __syncthreads();
    for (int k = 0; k < NK; ++k) {
        int cur = k & 1;
        if (k + 1 < NK) { gloadA(A, DCTX, (k + 1) * 32, tid, ar); gloadB32(W, DQ, (k + 1) * 32, n0, tid, br); }
        mma_step(At[cur], Bt[cur], wm, wn, lane, acc);
        if (k + 1 < NK) { __syncthreads(); storeA(At[1 - cur], tid, ar); storeB32t(Bt[1 - cur], tid, br); __syncthreads(); }
    }
    __bf16* Out = CW + (size_t)b * SCTX * DQ;
    const int mm0 = m0 + wm * 64, nn0 = n0 + wn * 64;
    const int n = lane & 15, mo = (lane >> 4) * 8;
#pragma unroll
    for (int i = 0; i < 4; ++i)
#pragma unroll
        for (int j = 0; j < 4; ++j)
#pragma unroll
            for (int r = 0; r < 8; ++r)
                Out[(size_t)(mm0 + i * 16 + mo + r) * DQ + nn0 + j * 16 + n] = tobf(acc[i][j][r]);
}

// ====== K2: attn[q,b,s] = query[q,b,:] . CW[b,s,:]  (+ mask) ==============
__global__ __launch_bounds__(256) void k2_scores(const float* __restrict__ qry,
                                                 const __bf16* __restrict__ CW,
                                                 const unsigned char* __restrict__ cmask,
                                                 float* __restrict__ attn) {
    __shared__ __attribute__((aligned(32))) __bf16 At[2][128 * 32];
    __shared__ __attribute__((aligned(32))) __bf16 Bt[2][256 * 32];
    const int tid = threadIdx.x, lane = tid & 31, wave = tid >> 5;
    const int wm = wave & 1, wn = wave >> 1;
    const int q0 = blockIdx.x * 128, s0 = blockIdx.y * 256, b = blockIdx.z;
    const float* A = qry + (size_t)q0 * BB * DQ + (size_t)b * DQ;        // ld = BB*DQ
    const __bf16* Bsrc = CW + (size_t)b * SCTX * DQ + (size_t)s0 * DQ;

    v8f acc[4][4] = {};
    float4 ar[4]; v8ui br[2];
    const int NK = DQ / 32;
    gloadA(A, BB * DQ, 0, tid, ar);  gloadBbf(Bsrc, DQ, 0, tid, br);
    storeA(At[0], tid, ar);          storeBbf(Bt[0], tid, br);
    __syncthreads();
    for (int k = 0; k < NK; ++k) {
        int cur = k & 1;
        if (k + 1 < NK) { gloadA(A, BB * DQ, (k + 1) * 32, tid, ar); gloadBbf(Bsrc, DQ, (k + 1) * 32, tid, br); }
        mma_step(At[cur], Bt[cur], wm, wn, lane, acc);
        if (k + 1 < NK) { __syncthreads(); storeA(At[1 - cur], tid, ar); storeBbf(Bt[1 - cur], tid, br); __syncthreads(); }
    }
    const int mm0 = q0 + wm * 64, nn0 = s0 + wn * 64;
    const int n = lane & 15, mo = (lane >> 4) * 8;
#pragma unroll
    for (int i = 0; i < 4; ++i)
#pragma unroll
        for (int j = 0; j < 4; ++j) {
            int sg = nn0 + j * 16 + n;
            bool m = cmask[(size_t)b * SCTX + sg] != 0;
#pragma unroll
            for (int r = 0; r < 8; ++r) {
                float v = m ? -1e30f : acc[i][j][r];
                attn[((size_t)(mm0 + i * 16 + mo + r) * BB + b) * SCTX + sg] = v;
            }
        }
}

// ====== K3: in-place softmax over s for each (q,b) row ====================
__global__ __launch_bounds__(256) void k3_softmax(float* __restrict__ attn) {
    float* row = attn + (size_t)blockIdx.x * SCTX;
    const int tid = threadIdx.x;
    __shared__ float red[256];
    float v[8], mx = -1e38f;
#pragma unroll
    for (int r = 0; r < 8; ++r) { v[r] = row[tid + 256 * r]; mx = fmaxf(mx, v[r]); }
    red[tid] = mx; __syncthreads();
    for (int s = 128; s > 0; s >>= 1) { if (tid < s) red[tid] = fmaxf(red[tid], red[tid + s]); __syncthreads(); }
    float rowmax = red[0]; __syncthreads();
    float sum = 0.f;
#pragma unroll
    for (int r = 0; r < 8; ++r) { v[r] = __expf(v[r] - rowmax); sum += v[r]; }
    red[tid] = sum; __syncthreads();
    for (int s = 128; s > 0; s >>= 1) { if (tid < s) red[tid] += red[tid + s]; __syncthreads(); }
    float scale = (rowmax <= -1e29f) ? 0.f : (1.f / red[0]);   // all-masked rows -> 0
#pragma unroll
    for (int r = 0; r < 8; ++r) row[tid + 256 * r] = v[r] * scale;
}

// ====== K4: comp[q,b,d] = sum_s attn[q,b,s] * context[b,s,d] ==============
__global__ __launch_bounds__(256) void k4_comp(const float* __restrict__ attn,
                                               const float* __restrict__ ctx,
                                               float* __restrict__ comp) {
    __shared__ __attribute__((aligned(32))) __bf16 At[2][128 * 32];
    __shared__ __attribute__((aligned(32))) __bf16 Bt[2][256 * 32];
    const int tid = threadIdx.x, lane = tid & 31, wave = tid >> 5;
    const int wm = wave & 1, wn = wave >> 1;
    const int q0 = blockIdx.x * 128, n0 = blockIdx.y * 256, b = blockIdx.z;
    const float* A = attn + (size_t)q0 * BB * SCTX + (size_t)b * SCTX;   // ld = BB*SCTX
    const float* Bsrc = ctx + (size_t)b * SCTX * DCTX;                   // [s][d], ld = DCTX

    v8f acc[4][4] = {};
    float4 ar[4], br[8];
    const int NK = SCTX / 32;
    gloadA(A, BB * SCTX, 0, tid, ar);  gloadB32(Bsrc, DCTX, 0, n0, tid, br);
    storeA(At[0], tid, ar);            storeB32t(Bt[0], tid, br);
    __syncthreads();
    for (int k = 0; k < NK; ++k) {
        int cur = k & 1;
        if (k + 1 < NK) { gloadA(A, BB * SCTX, (k + 1) * 32, tid, ar); gloadB32(Bsrc, DCTX, (k + 1) * 32, n0, tid, br); }
        mma_step(At[cur], Bt[cur], wm, wn, lane, acc);
        if (k + 1 < NK) { __syncthreads(); storeA(At[1 - cur], tid, ar); storeB32t(Bt[1 - cur], tid, br); __syncthreads(); }
    }
    const int mm0 = q0 + wm * 64, nn0 = n0 + wn * 64;
    const int n = lane & 15, mo = (lane >> 4) * 8;
#pragma unroll
    for (int i = 0; i < 4; ++i)
#pragma unroll
        for (int j = 0; j < 4; ++j)
#pragma unroll
            for (int r = 0; r < 8; ++r)
                comp[((size_t)(mm0 + i * 16 + mo + r) * BB + b) * DCTX + nn0 + j * 16 + n] = acc[i][j][r];
}

extern "C" void kernel_launch(void* const* d_in, const int* in_sizes, int n_in,
                              void* d_out, int out_size, void* d_ws, size_t ws_size,
                              hipStream_t stream) {
    const float* ctx          = (const float*)d_in[0];
    const float* qry          = (const float*)d_in[1];
    const float* W            = (const float*)d_in[2];
    const unsigned char* mask = (const unsigned char*)d_in[3];  // jax bool -> 1 byte

    float* attn = (float*)d_out;                                 // [SQ, BB, SCTX]
    float* comp = attn + (size_t)SQ * BB * SCTX;                 // [SQ, BB, DCTX]
    __bf16* CW  = (__bf16*)d_ws;                                 // bf16 [BB, SCTX, DQ] (64 MB)

    dim3 blk(256);
    k1_cw     <<<dim3(SCTX / 128, DQ   / 256, BB), blk, 0, stream>>>(ctx, W, CW);
    k2_scores <<<dim3(SQ   / 128, SCTX / 256, BB), blk, 0, stream>>>(qry, CW, mask, attn);
    k3_softmax<<<dim3(SQ * BB),                    blk, 0, stream>>>(attn);
    k4_comp   <<<dim3(SQ   / 128, DCTX / 256, BB), blk, 0, stream>>>(attn, ctx, comp);
}
